// VirtualNodeGatherAttn_37134287242009
// MI455X (gfx1250) — compile-verified, matrix operand order checked by reading, not run on hardware
//
#include <hip/hip_runtime.h>
#include <hip/hip_bf16.h>
#include <cmath>

// ---------------------------------------------------------------------------
// Problem constants (from reference)
// ---------------------------------------------------------------------------
#define N_NODES 16384
#define CS      384      // C_S (also HC = NUM_HEADS * C_ATTN = 384)
#define KVC     768      // 2 * HC
#define GRAPHS  32
#define VN      8
#define HEADS   8
#define CATT    48
#define NEG_INF_CONST 100000000.0f

typedef __attribute__((ext_vector_type(16))) __bf16 v16bf;
typedef __attribute__((ext_vector_type(8)))  float  v8f;
typedef __attribute__((ext_vector_type(2)))  float  v2f;
typedef __attribute__((ext_vector_type(4)))  unsigned int v4u;
typedef __attribute__((ext_vector_type(8)))  int    v8i;
typedef __attribute__((ext_vector_type(4)))  int    v4i;

union V16 {
    uint4        q[2];
    unsigned int u[8];
    v16bf        v;
};

// round-to-nearest-even f32 -> bf16 (bit pattern in low 16 bits)
__device__ __forceinline__ unsigned int f2bf1(float f) {
    unsigned int u = __float_as_uint(f);
    return (u + 0x7FFFu + ((u >> 16) & 1u)) >> 16;
}
__device__ __forceinline__ unsigned int pkbf(float lo, float hi) {
    return f2bf1(lo) | (f2bf1(hi) << 16);
}

// 16x32 bf16 fragment (ISA 16-bit A-matrix layout) from row-major bf16:
// lane (16*half + m) holds K = {k0+8h .. k0+8h+7, k0+16+8h .. k0+16+8h+7}
__device__ __forceinline__ v16bf frag_bf16(const unsigned short* __restrict__ row,
                                           int k0, int half) {
    V16 a;
    a.q[0] = *(const uint4*)(row + k0 + 8 * half);
    a.q[1] = *(const uint4*)(row + k0 + 16 + 8 * half);
    return a.v;
}

// ---------------------------------------------------------------------------
// Kernel 1: vectorized fp32 -> bf16 conversion (8 elems / thread, b128 I/O)
// ---------------------------------------------------------------------------
__global__ __launch_bounds__(256) void f32_to_bf16_vec(
    const float* __restrict__ src, unsigned short* __restrict__ dst, int n8) {
    int i = blockIdx.x * 256 + threadIdx.x;
    if (i < n8) {
        float4 a = ((const float4*)src)[2 * i];
        float4 b = ((const float4*)src)[2 * i + 1];
        uint4 o;
        o.x = pkbf(a.x, a.y); o.y = pkbf(a.z, a.w);
        o.z = pkbf(b.x, b.y); o.w = pkbf(b.z, b.w);
        ((uint4*)dst)[i] = o;
    }
}

// ---------------------------------------------------------------------------
// Kernel 2: KV projection GEMM  KV[N,768] = X[N,384] @ Wkv^T + bkv
// Block = 8 waves = 256 rows x 64 cols. The 48KB contiguous bf16 weight tile
// (64 rows x 384) is DMA'd into LDS with the Tensor Data Mover, then each
// wave runs 12 k-steps of 8 x v_wmma_f32_16x16x32_bf16 (A from global bf16,
// B from LDS).
// ---------------------------------------------------------------------------
__global__ __launch_bounds__(256) void kv_gemm_bf16(
    const unsigned short* __restrict__ Xb, const unsigned short* __restrict__ Wb,
    const float* __restrict__ bkv, float* __restrict__ KV) {

    __shared__ __attribute__((aligned(16))) unsigned short Bs[64 * CS]; // 48 KB

    const int blockRow = blockIdx.x / 12;
    const int nTile    = blockIdx.x - blockRow * 12;
    const int col0     = nTile * 64;
    const int row0     = blockRow * 256 + (threadIdx.x >> 5) * 32;
    const int lane     = threadIdx.x & 31;
    const int m        = lane & 15;
    const int half     = lane >> 4;

    // ---- stage B tile (contiguous 64*384 bf16 = 48KB) into LDS via TDM ----
#if __has_builtin(__builtin_amdgcn_tensor_load_to_lds)
    if (threadIdx.x < 32) {
        const unsigned long long gaddr =
            (unsigned long long)(const void*)(Wb + (size_t)col0 * CS);
        const unsigned int lds_addr = (unsigned int)(size_t)(const void*)Bs;
        const unsigned int elems = 64u * CS;          // 24576 bf16 elements

        // D# group0: count=1 | lds_addr | global_addr[56:0] | type=2
        v4u g0;
        g0.x = 1u;                                    // count = 1 valid descriptor
        g0.y = lds_addr;                              // LDS byte address
        g0.z = (unsigned int)gaddr;                   // global addr [31:0]
        g0.w = (unsigned int)((gaddr >> 32) & 0x01FFFFFFu) | (2u << 30); // +type=2

        // D# group1: wg_mask=0, data_size=1 (2B), 1-D tile:
        //  tensor_dim0 = tile_dim0 = 24576, tensor_dim1 = 1, stride0 = 24576
        v8i g1;
        g1[0] = (int)(1u << 16);                      // data_size = 2 bytes
        g1[1] = (int)((elems & 0xFFFFu) << 16);       // tensor_dim0[15:0] @bits63:48
        g1[2] = (int)(((elems >> 16) & 0xFFFFu) | (1u << 16)); // dim0 hi | tensor_dim1=1
        g1[3] = (int)((elems & 0xFFFFu) << 16);       // tensor_dim1 hi=0 | tile_dim0
        g1[4] = 0;                                    // tile_dim1 = tile_dim2 = 0
        g1[5] = (int)elems;                           // tensor_dim0_stride [31:0]
        g1[6] = 0;                                    // stride0 hi | stride1 lo
        g1[7] = 0;                                    // stride1 hi
        v4i g2 = {0, 0, 0, 0};
        v4i g3 = {0, 0, 0, 0};
#if defined(__clang_major__) && (__clang_major__ >= 23)
        v8i g1b = {0, 0, 0, 0, 0, 0, 0, 0};
        __builtin_amdgcn_tensor_load_to_lds(g0, g1, g2, g3, g1b, 0);
#else
        __builtin_amdgcn_tensor_load_to_lds(g0, g1, g2, g3, 0);
#endif
#if __has_builtin(__builtin_amdgcn_s_wait_tensorcnt)
        __builtin_amdgcn_s_wait_tensorcnt(0);
#endif
    }
#else
    // Fallback: cooperative copy global -> LDS
    {
        const uint4* src = (const uint4*)(Wb + (size_t)col0 * CS);
        uint4* dst = (uint4*)Bs;
        for (int i = threadIdx.x; i < (64 * CS) / 8; i += 256) dst[i] = src[i];
    }
#endif
    __syncthreads();

    // ---- main loop --------------------------------------------------------
    const unsigned short* arow0 = Xb + (size_t)(row0 + m) * CS;
    const unsigned short* arow1 = arow0 + 16 * CS;
    const unsigned short* brow0 = Bs + (size_t)m * CS;
    const unsigned short* brow1 = brow0 + 16 * CS;
    const unsigned short* brow2 = brow0 + 32 * CS;
    const unsigned short* brow3 = brow0 + 48 * CS;

    v8f acc[2][4] = {};

    for (int k0 = 0; k0 < CS; k0 += 32) {
        __builtin_prefetch(arow0 + k0 + 32, 0, 3);   // global_prefetch_b8 (WGP scope)
        v16bf a0 = frag_bf16(arow0, k0, half);
        v16bf a1 = frag_bf16(arow1, k0, half);
        v16bf b0 = frag_bf16(brow0, k0, half);       // ds_load_b128 x2
        v16bf b1 = frag_bf16(brow1, k0, half);
        v16bf b2 = frag_bf16(brow2, k0, half);
        v16bf b3 = frag_bf16(brow3, k0, half);

        acc[0][0] = __builtin_amdgcn_wmma_f32_16x16x32_bf16(false, a0, false, b0, (short)0, acc[0][0], false, false);
        acc[0][1] = __builtin_amdgcn_wmma_f32_16x16x32_bf16(false, a0, false, b1, (short)0, acc[0][1], false, false);
        acc[0][2] = __builtin_amdgcn_wmma_f32_16x16x32_bf16(false, a0, false, b2, (short)0, acc[0][2], false, false);
        acc[0][3] = __builtin_amdgcn_wmma_f32_16x16x32_bf16(false, a0, false, b3, (short)0, acc[0][3], false, false);
        acc[1][0] = __builtin_amdgcn_wmma_f32_16x16x32_bf16(false, a1, false, b0, (short)0, acc[1][0], false, false);
        acc[1][1] = __builtin_amdgcn_wmma_f32_16x16x32_bf16(false, a1, false, b1, (short)0, acc[1][1], false, false);
        acc[1][2] = __builtin_amdgcn_wmma_f32_16x16x32_bf16(false, a1, false, b2, (short)0, acc[1][2], false, false);
        acc[1][3] = __builtin_amdgcn_wmma_f32_16x16x32_bf16(false, a1, false, b3, (short)0, acc[1][3], false, false);
    }

    // D layout: VGPR r -> M = r + 8*half, N = lane&15
    #pragma unroll
    for (int j = 0; j < 4; ++j) {
        const int nn = col0 + 16 * j + m;
        const float bb = bkv[nn];
        #pragma unroll
        for (int i = 0; i < 2; ++i) {
            #pragma unroll
            for (int r = 0; r < 8; ++r) {
                const int mm = row0 + 16 * i + r + 8 * half;
                KV[(size_t)mm * KVC + nn] = acc[i][j][r] + bb;
            }
        }
    }
}

// ---------------------------------------------------------------------------
// Kernel 3: small GEMM  Out[M,Nc] = A[M,K] @ W[Nc,K]^T + bias
// Exact-precision fp32 WMMA (v_wmma_f32_16x16x4_f32), one wave per 16x16 tile.
// ---------------------------------------------------------------------------
__global__ __launch_bounds__(256) void gemm_wt_f32wmma(
    const float* __restrict__ A, const float* __restrict__ W,
    const float* __restrict__ bias, float* __restrict__ Out,
    int M, int Nc, int K) {

    const int wave   = (blockIdx.x << 3) + (threadIdx.x >> 5);
    const int lane   = threadIdx.x & 31;
    const int nTiles = Nc >> 4;
    const int mTile  = wave / nTiles;
    const int nTile  = wave - mTile * nTiles;
    if ((mTile << 4) >= M) return;          // uniform per wave
    const int m    = lane & 15;
    const int half = lane >> 4;

    const float* arow = A + (size_t)((mTile << 4) + m) * K;
    const float* wrow = W + (size_t)((nTile << 4) + m) * K;

    v8f acc = {};
    // f32 A 16x4: lanes 0-15 hold K = {k0, k0+1}, lanes 16-31 hold {k0+2, k0+3}
    for (int k0 = 0; k0 < K; k0 += 4) {
        float2 fa = *(const float2*)(arow + k0 + 2 * half);
        float2 fw = *(const float2*)(wrow + k0 + 2 * half);
        v2f av; av.x = fa.x; av.y = fa.y;
        v2f wv; wv.x = fw.x; wv.y = fw.y;
        acc = __builtin_amdgcn_wmma_f32_16x16x4_f32(false, av, false, wv, (short)0, acc, false, false);
    }

    const int nn = (nTile << 4) + m;
    const float bb = bias[nn];
    #pragma unroll
    for (int r = 0; r < 8; ++r) {
        const int mm = (mTile << 4) + r + 8 * half;
        Out[(size_t)mm * Nc + nn] = acc[r] + bb;
    }
}

// ---------------------------------------------------------------------------
// Kernel 4: fused per-graph flash softmax + weighted segment sum.
// batch is sorted -> node range per graph found by binary search.
// Faithful to reference: score *= (1-mask)*(-1e8); denom sums e over ALL
// nodes in segment; numerator weighted by mask.
// ---------------------------------------------------------------------------
__device__ __forceinline__ int lower_bound_i(const int* __restrict__ a, int n, int key) {
    int lo = 0, hi = n;
    while (lo < hi) { int mid = (lo + hi) >> 1; if (a[mid] < key) lo = mid + 1; else hi = mid; }
    return lo;
}

__global__ __launch_bounds__(256) void attn_flash(
    const float* __restrict__ KV, const float* __restrict__ vnq,
    const int* __restrict__ batch, const unsigned char* __restrict__ nmask,
    float* __restrict__ vnup, int Nn) {

    __shared__ float qs[VN * HEADS * CATT];   // 12 KB
    __shared__ float red[256 * 50];           // 50 KB: {m, s, acc[48]} per thread

    const int g = blockIdx.x;
    const int t = threadIdx.x;

    for (int i = t; i < VN * HEADS * CATT; i += 256)
        qs[i] = vnq[(size_t)g * VN * HEADS * CATT + i];

    const int lo = lower_bound_i(batch, Nn, g);
    const int hi = lower_bound_i(batch, Nn, g + 1);
    __syncthreads();

    const int vh = t & 63, stripe = t >> 6;
    const int h = vh & 7;
    const float* qrow = qs + vh * CATT;

    float mrun = -INFINITY, srun = 0.0f;
    float acc[CATT];
    #pragma unroll
    for (int c = 0; c < CATT; ++c) acc[c] = 0.0f;

    for (int n = lo + stripe; n < hi; n += 4) {
        const float4* k4 = (const float4*)(KV + (size_t)n * KVC + h * 96);
        float s = 0.0f;
        #pragma unroll
        for (int c4 = 0; c4 < 12; ++c4) {
            float4 kk = k4[c4];
            const float* qq = qrow + c4 * 4;
            s += kk.x * qq[0] + kk.y * qq[1] + kk.z * qq[2] + kk.w * qq[3];
        }
        s *= 0.14433756729740643f;                       // 1/sqrt(48)
        const float mf = nmask[n] ? 1.0f : 0.0f;
        s = s * ((1.0f - mf) * -NEG_INF_CONST);          // faithful to reference

        const float mnew  = fmaxf(mrun, s);
        const float scale = __expf(mrun - mnew);         // first iter: exp(-inf)=0
        const float e     = __expf(s - mnew);
        srun = srun * scale + e;
        const float w = e * mf;
        const float4* v4 = k4 + 12;
        #pragma unroll
        for (int c4 = 0; c4 < 12; ++c4) {
            float4 vv = v4[c4];
            acc[c4 * 4 + 0] = acc[c4 * 4 + 0] * scale + w * vv.x;
            acc[c4 * 4 + 1] = acc[c4 * 4 + 1] * scale + w * vv.y;
            acc[c4 * 4 + 2] = acc[c4 * 4 + 2] * scale + w * vv.z;
            acc[c4 * 4 + 3] = acc[c4 * 4 + 3] * scale + w * vv.w;
        }
        mrun = mnew;
    }

    float* slot = &red[t * 50];
    slot[0] = mrun; slot[1] = srun;
    #pragma unroll
    for (int c = 0; c < CATT; ++c) slot[2 + c] = acc[c];
    __syncthreads();

    if (t < 64) {
        float M = -INFINITY;
        #pragma unroll
        for (int p = 0; p < 4; ++p) M = fmaxf(M, red[(t + 64 * p) * 50]);
        float S = 0.0f;
        float out[CATT];
        #pragma unroll
        for (int c = 0; c < CATT; ++c) out[c] = 0.0f;
        #pragma unroll
        for (int p = 0; p < 4; ++p) {
            const float* sp = &red[(t + 64 * p) * 50];
            const float mi  = sp[0];
            const float wsc = (mi == -INFINITY) ? 0.0f : __expf(mi - M);
            S += sp[1] * wsc;
            #pragma unroll
            for (int c = 0; c < CATT; ++c) out[c] += sp[2 + c] * wsc;
        }
        const float inv = 1.0f / (S + 1e-16f);
        const int v = t >> 3, hh = t & 7;
        float* op = vnup + ((size_t)(g * VN + v)) * CS + hh * CATT;
        #pragma unroll
        for (int c = 0; c < CATT; ++c) op[c] = out[c] * inv;
    }
}

// ---------------------------------------------------------------------------
// Host-side launcher
// ---------------------------------------------------------------------------
extern "C" void kernel_launch(void* const* d_in, const int* in_sizes, int n_in,
                              void* d_out, int out_size, void* d_ws, size_t ws_size,
                              hipStream_t stream) {
    const float*         node_features = (const float*)d_in[0];
    const float*         vn_features   = (const float*)d_in[1];
    const int*           batch         = (const int*)d_in[2];
    const unsigned char* node_mask     = (const unsigned char*)d_in[3];
    const float*         Wq   = (const float*)d_in[4];
    const float*         bq   = (const float*)d_in[5];
    const float*         Wkv  = (const float*)d_in[6];
    const float*         bkv  = (const float*)d_in[7];
    const float*         Wout = (const float*)d_in[8];
    const float*         bout = (const float*)d_in[9];

    char* ws = (char*)d_ws;
    float* kv = (float*)ws;                     ws += (size_t)N_NODES * KVC * sizeof(float);     // 50.3 MB
    unsigned short* wkvb = (unsigned short*)ws; ws += (size_t)KVC * CS * sizeof(unsigned short); // 0.6 MB
    unsigned short* xb = (unsigned short*)ws;   ws += (size_t)N_NODES * CS * sizeof(unsigned short); // 12.6 MB
    float* vnq  = (float*)ws;                   ws += (size_t)GRAPHS * VN * CS * sizeof(float);  // 0.4 MB
    float* vnup = (float*)ws;                                                                    // 0.4 MB

    // 1) fp32 -> bf16: Wkv (once, L2-hot afterwards) and node_features
    f32_to_bf16_vec<<<(KVC * CS / 8 + 255) / 256, 256, 0, stream>>>(Wkv, wkvb, KVC * CS / 8);
    f32_to_bf16_vec<<<(N_NODES * CS / 8 + 255) / 256, 256, 0, stream>>>(node_features, xb, N_NODES * CS / 8);

    // 2) vn_q = vn_features @ Wq^T + bq   (256x384x384, exact fp32 WMMA)
    gemm_wt_f32wmma<<<((256 / 16) * (CS / 16) + 7) / 8, 256, 0, stream>>>(
        vn_features, Wq, bq, vnq, 256, CS, CS);

    // 3) kv = X_bf16 @ Wkv_bf16^T + bkv (16384x768x384, TDM-staged B + WMMA)
    kv_gemm_bf16<<<(N_NODES / 256) * 12, 256, 0, stream>>>(xb, wkvb, bkv, kv);

    // 4) fused flash softmax + weighted segment sum (one block per graph)
    attn_flash<<<GRAPHS, 256, 0, stream>>>(kv, vnq, batch, node_mask, vnup, N_NODES);

    // 5) out = vn_update @ Wout^T + bout  (256x384x384, exact fp32 WMMA)
    gemm_wt_f32wmma<<<((256 / 16) * (CS / 16) + 7) / 8, 256, 0, stream>>>(
        vnup, Wout, bout, (float*)d_out, 256, CS, CS);
}
